// MoERouter_25108378812434
// MI455X (gfx1250) — matrix operand order repaired and checked by reading, not compiled, not run on hardware
//
#include <hip/hip_runtime.h>
#include <hip/hip_bf16.h>
#include <cstdint>

typedef __attribute__((ext_vector_type(16))) __bf16 v16bf;
typedef __attribute__((ext_vector_type(8)))  __bf16 v8bf;
typedef __attribute__((ext_vector_type(8)))  float  v8f;
typedef __attribute__((ext_vector_type(4)))  float  v4f;
typedef __attribute__((ext_vector_type(4)))  unsigned int u32x4;
typedef __attribute__((ext_vector_type(8)))  int          i32x8;
typedef __attribute__((ext_vector_type(4)))  int          i32x4;

#define N_     16384
#define C_     2048
#define E_     64
#define TOPK   8
#define EPSF   1e-12f

#define KCHUNK     256                 // k-elements staged per TDM transfer
#define NCHUNK     (C_ / KCHUNK)       // 8
#define ROWPAD     8                   // +8 bf16 (16 B) per row, via TDM pad
#define ROWELT     (KCHUNK + ROWPAD)   // 264 bf16 per LDS row (528 B)
#define CHUNKELT   (E_ * ROWELT)       // 16896 bf16 per buffer
#define CHUNKBYTES (CHUNKELT * 2)      // 33792 B per buffer

// ---------------------------------------------------------------------------
// Prep: W [C,E] f32 (W[k*E+e]) -> Wt [E,C] bf16 (Wt[e*C+k]) in d_ws.
// ---------------------------------------------------------------------------
__global__ __launch_bounds__(256) void wt_prep_kernel(
    const float* __restrict__ W, __bf16* __restrict__ Wt) {
  int idx = blockIdx.x * 256 + threadIdx.x;     // 0 .. E_*C_-1
  int e = idx >> 11;
  int k = idx & (C_ - 1);
  Wt[idx] = (__bf16)W[(size_t)k * E_ + e];
}

// ---------------------------------------------------------------------------
// TDM: DMA one [KCHUNK x E_] bf16 tile of Wt into LDS, padding each
// 512 B row by 16 B (row stride 528 B -> conflict-free ds_load_b128 reads).
// 6-arg builtin on this toolchain: (g0, g1, g2, g3, g4, cpol).
// ---------------------------------------------------------------------------
__device__ __forceinline__ void tdm_load_chunk(const __bf16* gsrc,
                                               unsigned lds_off) {
  unsigned long long ga = (unsigned long long)(uintptr_t)gsrc;
  u32x4 g0;
  g0[0] = 1u;                                       // count=1 (valid, user mode)
  g0[1] = lds_off;                                  // lds_addr (bytes)
  g0[2] = (unsigned)(ga & 0xFFFFFFFFu);             // global_addr[31:0]
  g0[3] = (unsigned)((ga >> 32) & 0x01FFFFFFu)      // global_addr[56:32]
        | (2u << 30);                               // type = 2 ("image")
  i32x8 g1;
  g1[0] = (1 << 16)                                 // data_size = 1 -> 2 bytes
        | (1 << 20)                                 // pad_enable
        | (6 << 22)                                 // pad_interval: 128 DWORDs (512 B row)
        | (3 << 25);                                // pad_amount: 4 DWORDs (16 B)
  g1[1] = (int)((unsigned)(C_ & 0xFFFF) << 16);     // tensor_dim0[15:0] = 2048
  g1[2] = (int)((unsigned)(C_ >> 16) | (E_ << 16)); // tensor_dim0 hi | tensor_dim1 = 64
  g1[3] = (int)((unsigned)KCHUNK << 16);            // tile_dim0 = 256
  g1[4] = E_;                                       // tile_dim1 = 64, tile_dim2 = 0
  g1[5] = C_;                                       // tensor_dim0_stride = 2048
  g1[6] = 0;
  g1[7] = 0;
  i32x4 z4 = {0, 0, 0, 0};
  i32x8 z8 = {0, 0, 0, 0, 0, 0, 0, 0};
  __builtin_amdgcn_tensor_load_to_lds(g0, g1, z4, z4, z8, 0);
}

__device__ __forceinline__ v16bf ld_bfrag(const __bf16* p) {
  v8bf lo = *(const v8bf*)p;
  v8bf hi = *(const v8bf*)(p + 16);
  return __builtin_shufflevector(lo, hi, 0, 1, 2, 3, 4, 5, 6, 7,
                                 8, 9, 10, 11, 12, 13, 14, 15);
}

// ---------------------------------------------------------------------------
// Router: per wave, 16 rows x 64 experts via v_wmma_f32_16x16x32_bf16.
// W staged into LDS by the Tensor Data Mover, double buffered.
// ---------------------------------------------------------------------------
__global__ __launch_bounds__(256) void moe_router_kernel(
    const float*  __restrict__ x,      // [N_, C_]
    const __bf16* __restrict__ Wt,     // [E_, C_] bf16 (transposed W)
    const float*  __restrict__ bias,   // [E_]
    const float*  __restrict__ ebias,  // [E_]
    float*        __restrict__ out) {  // [N*K idx][N*K w][N*E probs]
  // 2 x 33792 B TDM double buffer; epilogue z-buffer (34816 B) aliases it
  // after the k-loop (dead by then). Total LDS = 67584 B.
  __shared__ __align__(1024) unsigned char smem[2 * CHUNKBYTES];

  const int lane = threadIdx.x & 31;
  const int wave = threadIdx.x >> 5;
  const int rowBase = (blockIdx.x * 8 + wave) * 16;

  // LDS byte offset of smem (low 32 bits of the generic pointer).
  const unsigned smemAddr = (unsigned)(uintptr_t)(void*)&smem[0];

  // A-fragment mapping (bf16 16x32): lane l -> row (l&15);
  // lanes 0-15 hold K 0-7 & 16-23, lanes 16-31 hold K 8-15 & 24-31.
  const int r  = lane & 15;
  const int kg = (lane >> 4) << 3;              // 0 or 8
  const float* xr = x + (size_t)(rowBase + r) * C_;

  // B-fragment mapping (bf16 32x16): lane l -> col (l&15);
  // lanes 0-15 hold K 0-7 & 16-23, lanes 16-31 hold K 8-15 & 24-31.
  const int laneOffB = (lane & 15) * ROWELT + ((lane >> 4) << 3);

  v8f acc0 = {}, acc1 = {}, acc2 = {}, acc3 = {};

  // Prologue: DMA chunk 0 into buffer 0.
  if (wave == 0) {
    tdm_load_chunk(Wt, smemAddr);
  }

  for (int c = 0; c < NCHUNK; ++c) {
    if (wave == 0) {
      if (c + 1 < NCHUNK) {
        // Overwrites the buffer last read in iteration c-1; the barrier at
        // the end of iteration c-1 guarantees all waves are done with it.
        tdm_load_chunk(Wt + (c + 1) * KCHUNK,
                       smemAddr + (unsigned)(((c + 1) & 1) * CHUNKBYTES));
        __builtin_amdgcn_s_wait_tensorcnt(1);   // chunk c landed in LDS
      } else {
        __builtin_amdgcn_s_wait_tensorcnt(0);
      }
    }
    __syncthreads();                            // chunk c visible to all waves

    const __bf16* bufc =
        (const __bf16*)(smem + (c & 1) * CHUNKBYTES) + laneOffB;
    const int ck = c * KCHUNK;

    #pragma unroll 2
    for (int kl = 0; kl < KCHUNK / 32; ++kl) {
      const int base = ck + kl * 32;

      // x streamed once: nontemporal, keep L2 for Wt.
      v4f f0 = __builtin_nontemporal_load((const v4f*)(xr + base + kg));
      v4f f1 = __builtin_nontemporal_load((const v4f*)(xr + base + kg + 4));
      v4f f2 = __builtin_nontemporal_load((const v4f*)(xr + base + 16 + kg));
      v4f f3 = __builtin_nontemporal_load((const v4f*)(xr + base + 16 + kg + 4));

      v16bf a;
      a[0]  = (__bf16)f0[0]; a[1]  = (__bf16)f0[1];
      a[2]  = (__bf16)f0[2]; a[3]  = (__bf16)f0[3];
      a[4]  = (__bf16)f1[0]; a[5]  = (__bf16)f1[1];
      a[6]  = (__bf16)f1[2]; a[7]  = (__bf16)f1[3];
      a[8]  = (__bf16)f2[0]; a[9]  = (__bf16)f2[1];
      a[10] = (__bf16)f2[2]; a[11] = (__bf16)f2[3];
      a[12] = (__bf16)f3[0]; a[13] = (__bf16)f3[1];
      a[14] = (__bf16)f3[2]; a[15] = (__bf16)f3[3];

      // Four expert column tiles from LDS (2x ds_load_b128 each,
      // bank-conflict-free thanks to the 528 B padded row stride).
      v16bf b0 = ld_bfrag(bufc + 0 * 16 * ROWELT + kl * 32);
      v16bf b1 = ld_bfrag(bufc + 1 * 16 * ROWELT + kl * 32);
      v16bf b2 = ld_bfrag(bufc + 2 * 16 * ROWELT + kl * 32);
      v16bf b3 = ld_bfrag(bufc + 3 * 16 * ROWELT + kl * 32);

      acc0 = __builtin_amdgcn_wmma_f32_16x16x32_bf16(false, a, false, b0,
                                                     (short)0, acc0, false, false);
      acc1 = __builtin_amdgcn_wmma_f32_16x16x32_bf16(false, a, false, b1,
                                                     (short)0, acc1, false, false);
      acc2 = __builtin_amdgcn_wmma_f32_16x16x32_bf16(false, a, false, b2,
                                                     (short)0, acc2, false, false);
      acc3 = __builtin_amdgcn_wmma_f32_16x16x32_bf16(false, a, false, b3,
                                                     (short)0, acc3, false, false);
    }
    __syncthreads();   // all waves done reading buffer (c&1) before reuse
  }

  // Epilogue z-buffer aliases the (now dead) TDM buffers.
  float* zbuf = (float*)smem;

  // C/D layout: VGPR v, lanes 0-15 -> (M=v, N=lane), lanes 16-31 -> (M=v+8).
  float* zw = zbuf + wave * (16 * 68);
  const int mOff  = (lane >> 4) << 3;
  const int eLane = lane & 15;
  #pragma unroll
  for (int v = 0; v < 8; ++v) {
    zw[(v + mOff) * 68 +  0 + eLane] = acc0[v];
    zw[(v + mOff) * 68 + 16 + eLane] = acc1[v];
    zw[(v + mOff) * 68 + 32 + eLane] = acc2[v];
    zw[(v + mOff) * 68 + 48 + eLane] = acc3[v];
  }
  __syncthreads();

  if (lane < 16) {
    float* zr = zw + lane * 68;          // 64 logits, stride 68: conflict-free
    const int grow = rowBase + lane;

    // Pass 1: s = sigmoid(z + b) + expert_bias; logit = log(max(s, eps))
    float m = -3.0e38f;
    for (int e = 0; e < E_; ++e) {
      float z = zr[e] + bias[e];
      float s = 1.0f / (1.0f + __expf(-z)) + ebias[e];
      float lg = __logf(fmaxf(s, EPSF));  // SCALING == 1.0
      zr[e] = lg;
      m = fmaxf(m, lg);
    }
    // Pass 2: softmax numerator + sum
    float sum = 0.0f;
    for (int e = 0; e < E_; ++e) {
      float p = __expf(zr[e] - m);
      sum += p;
      zr[e] = p;
    }
    float inv = 1.0f / sum;
    // Pass 3: probs_full out + keep in LDS for top-k
    float* probs = out + (size_t)2 * N_ * TOPK + (size_t)grow * E_;
    for (int e = 0; e < E_; ++e) {
      float pr = zr[e] * inv;
      probs[e] = pr;
      zr[e] = pr;
    }
    // Top-8 (first-occurrence ties via strict >), then renormalize
    float tw[TOPK]; int ti[TOPK];
    float ksum = 0.0f;
    #pragma unroll
    for (int j = 0; j < TOPK; ++j) {
      float best = -1.0f; int bi = 0;
      for (int e = 0; e < E_; ++e) {
        float v2 = zr[e];
        if (v2 > best) { best = v2; bi = e; }
      }
      zr[bi] = -1.0f;
      tw[j] = best; ti[j] = bi; ksum += best;
    }
    float dn = 1.0f / fmaxf(ksum, EPSF);
    #pragma unroll
    for (int j = 0; j < TOPK; ++j) {
      out[(size_t)grow * TOPK + j]                     = (float)ti[j];
      out[(size_t)N_ * TOPK + (size_t)grow * TOPK + j] = tw[j] * dn;
    }
  }
}

// ---------------------------------------------------------------------------
extern "C" void kernel_launch(void* const* d_in, const int* in_sizes, int n_in,
                              void* d_out, int out_size, void* d_ws, size_t ws_size,
                              hipStream_t stream) {
  const float* x  = (const float*)d_in[0];   // [N, C]
  const float* W  = (const float*)d_in[1];   // [C, E]
  const float* b  = (const float*)d_in[2];   // [E]
  const float* eb = (const float*)d_in[3];   // [E]
  float* out = (float*)d_out;
  __bf16* Wt = (__bf16*)d_ws;                // 64*2048*2 = 256 KB scratch

  wt_prep_kernel<<<dim3((C_ * E_) / 256), dim3(256), 0, stream>>>(W, Wt);
  moe_router_kernel<<<dim3(N_ / 128), dim3(256), 0, stream>>>(x, Wt, b, eb, out);
}
